// FragmentedLinear_64089501991435
// MI455X (gfx1250) — compile-verified, matrix operand order checked by reading, not run on hardware
//
#include <hip/hip_runtime.h>

// Problem sizes (fixed by the reference)
#define BATCH 8192
#define DIN   4096
#define NFRAG 32
#define FRAG  128
#define DOUT  4096
#define COMP  512

typedef float  v8f  __attribute__((ext_vector_type(8)));
typedef __bf16 v16bf __attribute__((ext_vector_type(16)));
typedef __bf16 v8bf  __attribute__((ext_vector_type(8)));
typedef __bf16 v4bf  __attribute__((ext_vector_type(4)));

// GEMM tiling: block tile 128(M) x 128(N), K-chunk 64, 256 threads = 8 waves
// in a 2(M) x 4(N) wave grid; each wave owns a 64x32 macro-tile = 4x2
// fp32 accumulators fed by V_WMMA_F32_16X16X32_BF16 (bf16x3 fp32 emulation).
// Double-buffered LDS with register prefetch: one barrier per K-chunk.
#define BM 128
#define BN 128
#define KC 64
#define LDK 72             // bf16 per LDS row (144 B: 16B-aligned, bank-perm)
#define TILE (128 * LDK)   // bf16 elements per tile array
#define SMEM_BYTES (2 * 4 * TILE * 2)  // 2 buffers x 4 arrays x bf16

// ---------------------------------------------------------------------------
// Kernel 1: fragment selector scores + softmax -> probs [BATCH, 32]
// One wave32 per batch row; lane == fragment index.
// ---------------------------------------------------------------------------
__global__ __launch_bounds__(256) void probs_kernel(
    const float* __restrict__ x,
    const float* __restrict__ selw,
    float* __restrict__ probs) {
  const int wave = (blockIdx.x * blockDim.x + threadIdx.x) >> 5;
  const int lane = threadIdx.x & 31;
  if (wave >= BATCH) return;

  const float4* xr = (const float4*)(x + (size_t)wave * DIN + lane * FRAG);
  const float4* wr = (const float4*)(selw + lane * FRAG);
  float acc = 0.f;
#pragma unroll 8
  for (int j = 0; j < FRAG / 4; ++j) {
    float4 a = xr[j];
    float4 b = wr[j];
    acc += a.x * b.x + a.y * b.y + a.z * b.z + a.w * b.w;
  }
  float m = acc;
#pragma unroll
  for (int off = 16; off; off >>= 1) m = fmaxf(m, __shfl_xor(m, off, 32));
  float e = __expf(acc - m);
  float s = e;
#pragma unroll
  for (int off = 16; off; off >>= 1) s += __shfl_xor(s, off, 32);
  probs[(size_t)wave * NFRAG + lane] = e / s;
}

// ---------------------------------------------------------------------------
// bf16 hi/lo split: x = hi + lo + O(2^-17 |x|)
// ---------------------------------------------------------------------------
struct bfpair {
  __bf16 h, l;
};

__device__ __forceinline__ bfpair split1(float x) {
  const float hf = __uint_as_float(__float_as_uint(x) & 0xffff0000u);
  bfpair r;
  r.h = (__bf16)hf;        // exact
  r.l = (__bf16)(x - hf);  // RNE of residual
  return r;
}

// ---------------------------------------------------------------------------
// Register prefetch payloads (32 VGPRs each)
// ---------------------------------------------------------------------------
struct PrefA { float4 v[8]; };  // BM x KC = 128x64 fp32 -> 8 float4/thread
struct PrefB { float4 v[8]; };  // KC x BN =  64x128 fp32 -> 8 float4/thread

// A tile global load, fused scaling. MODE: 0 plain, 1 *p, 2 *(1-p)
template <int MODE>
__device__ __forceinline__ void loadA_g(PrefA& pa,
                                        const float* __restrict__ A,
                                        const float* __restrict__ probs,
                                        int bm0, int k0, int lda, int tid) {
#pragma unroll
  for (int i = 0; i < 8; ++i) {
    const int q = tid + i * 256;  // 2048 float4
    const int row = q >> 4;       // 16 float4 per 64-float row
    const int c4 = (q & 15) << 2;
    float4 v = *(const float4*)(A + (size_t)(bm0 + row) * lda + k0 + c4);
    if (MODE != 0) {
      const float p = probs[(size_t)(bm0 + row) * NFRAG + (k0 >> 7)];
      const float sc = (MODE == 1) ? p : (1.f - p);
      v.x *= sc; v.y *= sc; v.z *= sc; v.w *= sc;
    }
    pa.v[i] = v;
  }
}

// A tile split + LDS store, layout [m][k] bf16 (row stride LDK)
__device__ __forceinline__ void storeA_s(__bf16* __restrict__ Ah,
                                         __bf16* __restrict__ Al,
                                         const PrefA& pa, int tid) {
#pragma unroll
  for (int i = 0; i < 8; ++i) {
    const int q = tid + i * 256;
    const int row = q >> 4;
    const int c4 = (q & 15) << 2;
    const float4 v = pa.v[i];
    const bfpair px = split1(v.x), py = split1(v.y);
    const bfpair pz = split1(v.z), pw = split1(v.w);
    v4bf h, l;
    h.x = px.h; h.y = py.h; h.z = pz.h; h.w = pw.h;
    l.x = px.l; l.y = py.l; l.z = pz.l; l.w = pw.l;
    *(v4bf*)&Ah[row * LDK + c4] = h;
    *(v4bf*)&Al[row * LDK + c4] = l;
  }
}

// B tile global load: (k-quad, n-quad) groups; 2 groups x 4 rows of float4
__device__ __forceinline__ void loadB_g(PrefB& pb,
                                        const float* __restrict__ Bsrc,
                                        int k0, int n0, int ldb, int tid) {
#pragma unroll
  for (int i = 0; i < 2; ++i) {
    const int g = tid + i * 256;  // 512 groups = 16 kq x 32 nq
    const int nq = g & 31;
    const int kq = g >> 5;
    const float* p0 = Bsrc + (size_t)(k0 + 4 * kq) * ldb + n0 + 4 * nq;
#pragma unroll
    for (int rr = 0; rr < 4; ++rr)
      pb.v[i * 4 + rr] = *(const float4*)(p0 + (size_t)rr * ldb);
  }
}

// B tile split + transposed LDS store: [n][k] bf16, b64 stores (4 k per col)
__device__ __forceinline__ void storeB_s(__bf16* __restrict__ Bh,
                                         __bf16* __restrict__ Bl,
                                         const PrefB& pb, int tid) {
#pragma unroll
  for (int i = 0; i < 2; ++i) {
    const int g = tid + i * 256;
    const int nq = g & 31;
    const int kq = g >> 5;
#pragma unroll
    for (int j = 0; j < 4; ++j) {  // column within the quad
      v4bf h, l;
#pragma unroll
      for (int rr = 0; rr < 4; ++rr) {
        const bfpair p = split1((&pb.v[i * 4 + rr].x)[j]);
        ((__bf16*)&h)[rr] = p.h;
        ((__bf16*)&l)[rr] = p.l;
      }
      *(v4bf*)&Bh[(4 * nq + j) * LDK + 4 * kq] = h;
      *(v4bf*)&Bl[(4 * nq + j) * LDK + 4 * kq] = l;
    }
  }
}

// ---------------------------------------------------------------------------
// Fragment gather: per-lane K runs [ks + h*8, +8) and [ks + 16 + h*8, +8)
// (16-bit operand layout, ISA 7.12.2); each run one ds_load_b128.
// ---------------------------------------------------------------------------
__device__ __forceinline__ v16bf load_frag(const __bf16* __restrict__ T,
                                           int row, int ks, int h) {
  const v8bf r0 = *(const v8bf*)&T[row * LDK + ks + h * 8];
  const v8bf r1 = *(const v8bf*)&T[row * LDK + ks + 16 + h * 8];
  return __builtin_shufflevector(r0, r1, 0, 1, 2, 3, 4, 5, 6, 7, 8, 9, 10, 11,
                                 12, 13, 14, 15);
}

// ---------------------------------------------------------------------------
// One K-chunk of bf16x3 WMMA: C += Ah@Bh + Ah@Bl + Al@Bh
// B fragments cached per ks step; A fragments streamed per mi to cap liveness.
// ---------------------------------------------------------------------------
__device__ __forceinline__ void mma_chunk(const __bf16* __restrict__ Ah,
                                          const __bf16* __restrict__ Al,
                                          const __bf16* __restrict__ Bh,
                                          const __bf16* __restrict__ Bl,
                                          int wm, int wn, int lane,
                                          v8f c[4][2]) {
  const int r = lane & 15;
  const int h = lane >> 4;
#pragma unroll
  for (int ks = 0; ks < KC; ks += 32) {
    v16bf bh[2], bl[2];
#pragma unroll
    for (int ni = 0; ni < 2; ++ni) {
      const int n = wn * 32 + ni * 16 + r;
      bh[ni] = load_frag(Bh, n, ks, h);
      bl[ni] = load_frag(Bl, n, ks, h);
    }
#pragma unroll
    for (int mi = 0; mi < 4; ++mi) {
      const int m = wm * 64 + mi * 16 + r;
      const v16bf ahm = load_frag(Ah, m, ks, h);
      const v16bf alm = load_frag(Al, m, ks, h);
#pragma unroll
      for (int ni = 0; ni < 2; ++ni) {
        c[mi][ni] = __builtin_amdgcn_wmma_f32_16x16x32_bf16(
            false, ahm, false, bh[ni], (short)0, c[mi][ni], false, false);
        c[mi][ni] = __builtin_amdgcn_wmma_f32_16x16x32_bf16(
            false, ahm, false, bl[ni], (short)0, c[mi][ni], false, false);
        c[mi][ni] = __builtin_amdgcn_wmma_f32_16x16x32_bf16(
            false, alm, false, bh[ni], (short)0, c[mi][ni], false, false);
      }
    }
  }
}

// ---------------------------------------------------------------------------
// Double-buffered GEMM pass over one (A, B) pair, accumulating into c[][]
// ---------------------------------------------------------------------------
template <int MODE>
__device__ __forceinline__ void gemm_pass(__bf16* __restrict__ sm,
                                          const float* __restrict__ A,
                                          const float* __restrict__ probs,
                                          const float* __restrict__ Bsrc,
                                          int lda, int ldb, int K, int bm0,
                                          int n0, int tid, int lane, int wm,
                                          int wn, v8f c[4][2]) {
  PrefA pa;
  PrefB pb;
  loadA_g<MODE>(pa, A, probs, bm0, 0, lda, tid);
  loadB_g(pb, Bsrc, 0, n0, ldb, tid);
  __syncthreads();  // protect readers of the previous pass / prior contents
  storeA_s(sm, sm + TILE, pa, tid);
  storeB_s(sm + 2 * TILE, sm + 3 * TILE, pb, tid);

  int cur = 0;
  for (int k0 = 0; k0 < K; k0 += KC) {
    __syncthreads();
    const bool notlast = (k0 + KC) < K;
    if (notlast) {
      loadA_g<MODE>(pa, A, probs, bm0, k0 + KC, lda, tid);
      loadB_g(pb, Bsrc, k0 + KC, n0, ldb, tid);
    }
    __bf16* buf = sm + cur * 4 * TILE;
    mma_chunk(buf, buf + TILE, buf + 2 * TILE, buf + 3 * TILE, wm, wn, lane, c);
    if (notlast) {
      __bf16* nbuf = sm + (cur ^ 1) * 4 * TILE;
      storeA_s(nbuf, nbuf + TILE, pa, tid);
      storeB_s(nbuf + 2 * TILE, nbuf + 3 * TILE, pb, tid);
      cur ^= 1;
    }
  }
}

__device__ __forceinline__ void store_C(float* __restrict__ C, int ldc,
                                        int bm0, int n0, int wm, int wn,
                                        int lane, v8f c[4][2]) {
  const int col = lane & 15;
  const int hi = lane >> 4;
#pragma unroll
  for (int mi = 0; mi < 4; ++mi)
#pragma unroll
    for (int ni = 0; ni < 2; ++ni)
#pragma unroll
      for (int r = 0; r < 8; ++r) {
        const int gm = bm0 + wm * 64 + mi * 16 + r + 8 * hi;
        const int gn = n0 + wn * 32 + ni * 16 + col;
        C[(size_t)gm * ldc + gn] = c[mi][ni][r];
      }
}

// ---------------------------------------------------------------------------
// Kernel 2: T = (x * (1-p_rep)) @ Wc1   [8192,4096]x[4096,512] -> [8192,512]
// ---------------------------------------------------------------------------
__global__ __launch_bounds__(256) void compress1_kernel(
    const float* __restrict__ x, const float* __restrict__ probs,
    const float* __restrict__ Wc1, float* __restrict__ T) {
  extern __shared__ __align__(16) char smem_raw[];
  __bf16* sm = (__bf16*)smem_raw;
  const int tid = threadIdx.x;
  const int lane = tid & 31;
  const int wave = tid >> 5;
  const int wm = wave >> 2, wn = wave & 3;
  const int bm0 = blockIdx.y * BM;
  const int n0 = blockIdx.x * BN;

  v8f c[4][2] = {};
  gemm_pass<2>(sm, x, probs, Wc1, DIN, COMP, DIN, bm0, n0, tid, lane, wm, wn, c);
  store_C(T, COMP, bm0, n0, wm, wn, lane, c);
}

// ---------------------------------------------------------------------------
// Kernel 3: out = (x * p_rep) @ We  +  T @ Wc2   (fused accumulate)
// ---------------------------------------------------------------------------
__global__ __launch_bounds__(256) void moe_out_kernel(
    const float* __restrict__ x, const float* __restrict__ probs,
    const float* __restrict__ We, const float* __restrict__ T,
    const float* __restrict__ Wc2, float* __restrict__ out) {
  extern __shared__ __align__(16) char smem_raw[];
  __bf16* sm = (__bf16*)smem_raw;
  const int tid = threadIdx.x;
  const int lane = tid & 31;
  const int wave = tid >> 5;
  const int wm = wave >> 2, wn = wave & 3;
  const int bm0 = blockIdx.y * BM;
  const int n0 = blockIdx.x * BN;

  v8f c[4][2] = {};
  // Pass 1: expert path, K = 4096, A = x * p
  gemm_pass<1>(sm, x, probs, We, DIN, DOUT, DIN, bm0, n0, tid, lane, wm, wn, c);
  // Pass 2: compressed path, K = 512, A = T
  gemm_pass<0>(sm, T, nullptr, Wc2, COMP, DOUT, COMP, bm0, n0, tid, lane, wm,
               wn, c);
  store_C(out, DOUT, bm0, n0, wm, wn, lane, c);
}

// ---------------------------------------------------------------------------
// Launch
// ---------------------------------------------------------------------------
extern "C" void kernel_launch(void* const* d_in, const int* in_sizes, int n_in,
                              void* d_out, int out_size, void* d_ws,
                              size_t ws_size, hipStream_t stream) {
  const float* x    = (const float*)d_in[0];
  const float* selw = (const float*)d_in[1];
  const float* We   = (const float*)d_in[2];  // [32,128,4096] == [4096,4096]
  const float* Wc1  = (const float*)d_in[3];  // [4096,512]
  const float* Wc2  = (const float*)d_in[4];  // [512,4096]
  float* out = (float*)d_out;

  float* probs = (float*)d_ws;                                  // 1 MB
  float* T = (float*)((char*)d_ws + (size_t)BATCH * NFRAG * 4); // 16 MB

  probs_kernel<<<BATCH / 8, 256, 0, stream>>>(x, selw, probs);

  dim3 g2(COMP / BN, BATCH / BM);  // (4, 64)
  compress1_kernel<<<g2, 256, SMEM_BYTES, stream>>>(x, probs, Wc1, T);

  dim3 g3(DOUT / BN, BATCH / BM);  // (32, 64)
  moe_out_kernel<<<g3, 256, SMEM_BYTES, stream>>>(x, probs, We, T, Wc2, out);
}
